// TransformerBlockQuantum_65481071409199
// MI455X (gfx1250) — matrix operand order repaired and checked by reading, not compiled, not run on hardware
//
#include <hip/hip_runtime.h>
#include <hip/hip_bf16.h>

// ---------------------------------------------------------------- types ----
typedef __attribute__((ext_vector_type(16))) __bf16 bf16x16;
typedef __attribute__((ext_vector_type(8)))  __bf16 bf16x8;
typedef __attribute__((ext_vector_type(8)))  float  v8f;
typedef __attribute__((ext_vector_type(4)))  unsigned int u32x4;
typedef __attribute__((ext_vector_type(8)))  int i32x8;
typedef __attribute__((ext_vector_type(4)))  int i32x4;

#if defined(__AMDGCN__) && __has_builtin(__builtin_amdgcn_tensor_load_to_lds)
#define USE_TDM 1
#endif

union AFragU {
    bf16x16 v;
    uint4   u[2];
};

__device__ inline v8f zero8() {
    v8f z;
#pragma unroll
    for (int i = 0; i < 8; ++i) z[i] = 0.0f;
    return z;
}

__device__ inline v8f wmma_bf16(const AFragU& a, const AFragU& b, v8f c) {
    // D = A(16x32 bf16) * B(32x16 bf16) + C(16x16 f32)
    return __builtin_amdgcn_wmma_f32_16x16x32_bf16(
        /*neg_a=*/false, a.v, /*neg_b=*/false, b.v,
        /*c_mod=*/(short)0, c, /*reuse_a=*/false, /*reuse_b=*/false);
}

// ------------------------------------------------------ problem constants --
#define CB    2
#define CS    2048
#define CE    1024
#define CH    32
#define CDK   32
#define CFFN  4096
#define CM    (CB * CS)   // 4096 rows in all GEMMs

// --------------------------------------------------------------- convert ---
__global__ __launch_bounds__(256) void cvt_f32_bf16_kernel(
    const float* __restrict__ in, __bf16* __restrict__ out, int n) {
    int i = (blockIdx.x * 256 + threadIdx.x) * 4;
    if (i + 3 < n) {
        float4 v = *(const float4*)(in + i);
        union { __bf16 b[4]; uint2 u; } o;
        o.b[0] = (__bf16)v.x; o.b[1] = (__bf16)v.y;
        o.b[2] = (__bf16)v.z; o.b[3] = (__bf16)v.w;
        *(uint2*)(out + i) = o.u;
    }
}

// ------------------------------------------------------------------ GEMM ---
// C[M,N] = A[M,K] * B[K,N] (+bias) with selectable epilogue.
// Block tile 128x128, K-step 32, 256 threads = 8 waves, wave grid 4x2,
// each wave computes a 32x64 sub-tile as 2x4 WMMA 16x16 tiles.
// Double-buffered pipeline: the Tensor Data Mover stages the NEXT A tile
// (with per-row padding: 64B data + 16B pad -> 80B stride matching
// As[][BK+8]) and the next B tile's global loads are in flight while the
// current stage's WMMAs execute.
enum Epi { EPI_F32 = 0, EPI_BF16_RELU = 1, EPI_QKV = 2, EPI_QKV_SCALED = 3, EPI_VT = 4 };

template <int EPI>
__global__ __launch_bounds__(256) void gemm_bf16_kernel(
    const __bf16* __restrict__ A, const __bf16* __restrict__ Bm,
    const float* __restrict__ bias, const float* __restrict__ scale,
    void* __restrict__ out, int M, int N, int K, int Sseq) {
    constexpr int BM = 128, BN = 128, BK = 32, PAD = 8;
    __shared__ __attribute__((aligned(16))) __bf16 As[2][BM][BK + PAD];
    __shared__ __attribute__((aligned(16))) __bf16 Bt[2][BN][BK + PAD];

    const int tid  = threadIdx.x;
    const int lane = tid & 31;
    const int w    = tid >> 5;
    const int wm   = w >> 1;        // 0..3
    const int wn   = w & 1;         // 0..1
    const int half = lane >> 4;     // 0/1
    const int l16  = lane & 15;

    const int bm = blockIdx.y * BM;
    const int bn = blockIdx.x * BN;

    // ---- stage helpers ----------------------------------------------------
    auto issueA = [&](int kk, int bsel) {
#if USE_TDM
        if (w == 0) {  // one DMA per block; descriptors are block-uniform
            unsigned long long ga =
                (unsigned long long)(A + (size_t)bm * K + kk);
            unsigned int lds =
                (unsigned int)(unsigned long long)&As[bsel][0][0];
            u32x4 g0;
            g0[0] = 1u;                                   // count=1, user D#
            g0[1] = lds;                                  // lds_addr
            g0[2] = (unsigned int)ga;                     // global_addr[31:0]
            g0[3] = (unsigned int)((ga >> 32) & 0x1FFFFFFu) | (2u << 30); // type=2
            i32x8 g1;
            // data_size=1(2B) | pad_enable | pad_interval=3(16 dwords) |
            // pad_amount=3(4 dwords = 16B)
            g1[0] = (int)((1u << 16) | (1u << 20) | (3u << 22) | (3u << 25));
            g1[1] = (int)(((unsigned)K & 0xffffu) << 16);             // dim0 lo16
            g1[2] = (int)((((unsigned)K >> 16) & 0xffffu) |
                          (((unsigned)M & 0xffffu) << 16));           // dim0 hi/dim1 lo
            g1[3] = (int)((((unsigned)M >> 16) & 0xffffu) |
                          ((unsigned)BK << 16));                      // tile_dim0=32
            g1[4] = (int)(unsigned)BM;                                // tile_dim1=128
            g1[5] = (int)(unsigned)K;                                 // dim0_stride lo
            g1[6] = 0;
            g1[7] = 0;
            i32x4 gz = {0, 0, 0, 0};
#if __clang_major__ >= 23
            i32x8 gz8 = {0, 0, 0, 0, 0, 0, 0, 0};
            __builtin_amdgcn_tensor_load_to_lds(g0, g1, gz, gz, gz8, 0);
#else
            __builtin_amdgcn_tensor_load_to_lds(g0, g1, gz, gz, 0);
#endif
        }
#else
        // fallback: stage A tile through VGPRs
        int arow = tid >> 1;
        int acol = (tid & 1) * 16;
        const uint4* src =
            (const uint4*)(A + (size_t)(bm + arow) * K + kk + acol);
        uint4 v0 = src[0];
        uint4 v1 = src[1];
        *(uint4*)&As[bsel][arow][acol]     = v0;
        *(uint4*)&As[bsel][arow][acol + 8] = v1;
#endif
    };
    auto waitA = [&]() {
#if USE_TDM
        if (w == 0) __builtin_amdgcn_s_wait_tensorcnt(0);
#endif
    };

    bf16x8 bv[2];
    auto loadB = [&](int kk) {  // global -> registers (in flight over compute)
#pragma unroll
        for (int rep = 0; rep < 2; ++rep) {
            int u  = tid + rep * 256;
            int kr = u >> 4;
            int nc = (u & 15) * 8;
            bv[rep] = *(const bf16x8*)(Bm + (size_t)(kk + kr) * N + bn + nc);
        }
    };
    auto storeB = [&](int bsel) {  // registers -> transposed LDS tile
#pragma unroll
        for (int rep = 0; rep < 2; ++rep) {
            int u  = tid + rep * 256;
            int kr = u >> 4;
            int nc = (u & 15) * 8;
#pragma unroll
            for (int i = 0; i < 8; ++i) Bt[bsel][nc + i][kr] = bv[rep][i];
        }
    };

    v8f acc[2][4];
#pragma unroll
    for (int mt = 0; mt < 2; ++mt)
#pragma unroll
        for (int nt = 0; nt < 4; ++nt) acc[mt][nt] = zero8();

    // ---- pipeline prologue: stage k0 = 0 into buffer 0 --------------------
    issueA(0, 0);
    loadB(0);
    storeB(0);
    waitA();
    __syncthreads();

    int buf = 0;
    for (int k0 = 0; k0 < K; k0 += BK) {
        const int nxt = k0 + BK;
        if (nxt < K) {
            issueA(nxt, buf ^ 1);  // TDM runs during this stage's WMMAs
            loadB(nxt);            // B global loads in flight over compute
            if (nxt + BK < K)      // L2 prefetch for the tile after next
                __builtin_prefetch(
                    Bm + (size_t)(nxt + BK + (tid >> 4)) * N + bn, 0, 1);
        }

        AFragU af[2];
#pragma unroll
        for (int mt = 0; mt < 2; ++mt) {
            int row = wm * 32 + mt * 16 + l16;
            af[mt].u[0] = *(const uint4*)&As[buf][row][half * 8];
            af[mt].u[1] = *(const uint4*)&As[buf][row][16 + half * 8];
        }
#pragma unroll
        for (int nt = 0; nt < 4; ++nt) {
            int col = wn * 64 + nt * 16 + l16;
            AFragU bfrag;
            bfrag.u[0] = *(const uint4*)&Bt[buf][col][half * 8];
            bfrag.u[1] = *(const uint4*)&Bt[buf][col][16 + half * 8];
#pragma unroll
            for (int mt = 0; mt < 2; ++mt)
                acc[mt][nt] = wmma_bf16(af[mt], bfrag, acc[mt][nt]);
        }

        if (nxt < K) {
            storeB(buf ^ 1);  // publish next B tile
            waitA();          // next A tile DMA complete
        }
        __syncthreads();
        buf ^= 1;
    }

    // ------------------------------- epilogue -------------------------------
#pragma unroll
    for (int mt = 0; mt < 2; ++mt) {
#pragma unroll
        for (int nt = 0; nt < 4; ++nt) {
            int col    = bn + wn * 64 + nt * 16 + l16;
            float bcol = bias ? bias[col] : 0.0f;
#pragma unroll
            for (int j = 0; j < 8; ++j) {
                int   row = bm + wm * 32 + mt * 16 + half * 8 + j;
                float val = acc[mt][nt][j] + bcol;
                if constexpr (EPI == EPI_F32) {
                    ((float*)out)[(size_t)row * N + col] = val;
                } else if constexpr (EPI == EPI_BF16_RELU) {
                    val = fmaxf(val, 0.0f);
                    ((__bf16*)out)[(size_t)row * N + col] = (__bf16)val;
                } else if constexpr (EPI == EPI_QKV || EPI == EPI_QKV_SCALED) {
                    int h = col >> 5, dk = col & 31;
                    if constexpr (EPI == EPI_QKV_SCALED)
                        val *= scale[dk * CH + h];  // fold kernelW[d,h] into Q
                    int b = row / Sseq, s = row % Sseq;
                    ((__bf16*)out)[(((size_t)(b * CH + h)) * Sseq + s) * CDK + dk] =
                        (__bf16)val;
                } else {  // EPI_VT : V transposed per head [BH][DK][S]
                    int h = col >> 5, dk = col & 31;
                    int b = row / Sseq, s = row % Sseq;
                    ((__bf16*)out)[(((size_t)(b * CH + h)) * CDK + dk) * Sseq + s] =
                        (__bf16)val;
                }
            }
        }
    }
}

// ----------------------------------------------------------- attention -----
// Flash attention per (b,h). Block: 256 threads = 8 waves; wave owns 16 query
// rows; block covers 128 q rows. Scores & PV via WMMA bf16.
__global__ __launch_bounds__(256) void attn_kernel(
    const __bf16* __restrict__ Q, const __bf16* __restrict__ Kb,
    const __bf16* __restrict__ Vt, __bf16* __restrict__ O, int S) {
    __shared__ float P[8][16][33];  // wave-private P tiles (f32 -> bf16 relayout)

    const int tid  = threadIdx.x;
    const int lane = tid & 31;
    const int w    = tid >> 5;
    const int half = lane >> 4;
    const int l16  = lane & 15;

    const int qchunks = S / 128;
    const int bh      = blockIdx.x / qchunks;
    const int qblk    = blockIdx.x % qchunks;
    const int b       = bh >> 5;
    const int h       = bh & 31;
    const int qbase   = qblk * 128 + w * 16;

    const size_t hbase = (size_t)bh * S * CDK;

    // Q A-fragment (rows contiguous in d) — Q already scaled by kernelW[:,h]
    AFragU qf;
    {
        const __bf16* p = Q + hbase + (size_t)(qbase + l16) * CDK;
        qf.u[0] = *(const uint4*)(p + half * 8);
        qf.u[1] = *(const uint4*)(p + 16 + half * 8);
    }

    v8f o0 = zero8(), o1 = zero8();
    float m_i[8], l_i[8];
#pragma unroll
    for (int r = 0; r < 8; ++r) { m_i[r] = -1e30f; l_i[r] = 0.0f; }

    for (int kb = 0; kb < S; kb += 32) {
        // K B-fragments for two key halves (keys contiguous in d)
        AFragU kf0, kf1;
        {
            const __bf16* p0 = Kb + hbase + (size_t)(kb + l16) * CDK + half * 8;
            kf0.u[0] = *(const uint4*)p0;
            kf0.u[1] = *(const uint4*)(p0 + 16);
            const __bf16* p1 =
                Kb + hbase + (size_t)(kb + 16 + l16) * CDK + half * 8;
            kf1.u[0] = *(const uint4*)p1;
            kf1.u[1] = *(const uint4*)(p1 + 16);
        }
        v8f s0 = wmma_bf16(qf, kf0, zero8());
        v8f s1 = wmma_bf16(qf, kf1, zero8());

        // online softmax: row stats via width-16 cross-lane reductions
        float p0v[8], p1v[8];
#pragma unroll
        for (int r = 0; r < 8; ++r) {
            float t = fmaxf(s0[r], s1[r]);
#pragma unroll
            for (int ms = 1; ms < 16; ms <<= 1) t = fmaxf(t, __shfl_xor(t, ms, 16));
            float mnew = fmaxf(m_i[r], t);
            float corr = __expf(m_i[r] - mnew);
            float e0   = __expf(s0[r] - mnew);
            float e1   = __expf(s1[r] - mnew);
            float rs   = e0 + e1;
#pragma unroll
            for (int ms = 1; ms < 16; ms <<= 1) rs += __shfl_xor(rs, ms, 16);
            l_i[r] = l_i[r] * corr + rs;
            m_i[r] = mnew;
            o0[r] *= corr;
            o1[r] *= corr;
            p0v[r] = e0;
            p1v[r] = e1;
        }

        // relayout P (C layout -> A fragment) through wave-private LDS
#pragma unroll
        for (int r = 0; r < 8; ++r) {
            int prow = half * 8 + r;
            P[w][prow][l16]      = p0v[r];
            P[w][prow][16 + l16] = p1v[r];
        }
        asm volatile("s_wait_dscnt 0" ::: "memory");
        AFragU pf;
        {
            const float* pr = &P[w][l16][0];
#pragma unroll
            for (int i = 0; i < 8; ++i) pf.v[i] = (__bf16)pr[half * 8 + i];
#pragma unroll
            for (int i = 0; i < 8; ++i) pf.v[8 + i] = (__bf16)pr[16 + half * 8 + i];
        }

        // V B-fragments from V^T (keys contiguous per dk row)
        AFragU vf0, vf1;
        {
            const __bf16* vb = Vt + (size_t)bh * CDK * S;
            const __bf16* p0 = vb + (size_t)l16 * S + kb + half * 8;
            vf0.u[0] = *(const uint4*)p0;
            vf0.u[1] = *(const uint4*)(p0 + 16);
            const __bf16* p1 = vb + (size_t)(16 + l16) * S + kb + half * 8;
            vf1.u[0] = *(const uint4*)p1;
            vf1.u[1] = *(const uint4*)(p1 + 16);
        }
        o0 = wmma_bf16(pf, vf0, o0);
        o1 = wmma_bf16(pf, vf1, o1);
    }

    // normalize and write O in [B,S,E] bf16 layout
#pragma unroll
    for (int r = 0; r < 8; ++r) {
        float inv = 1.0f / l_i[r];
        int   s   = qbase + half * 8 + r;
        size_t ro = ((size_t)b * S + s) * CE + h * CDK;
        O[ro + l16]      = (__bf16)(o0[r] * inv);
        O[ro + 16 + l16] = (__bf16)(o1[r] * inv);
    }
}

// --------------------------------------------------- residual + layernorm --
template <bool WRITE_BF16>
__global__ __launch_bounds__(256) void resid_ln_kernel(
    const float* __restrict__ X, const float* __restrict__ T,
    const float* __restrict__ g, const float* __restrict__ be,
    float* __restrict__ Yf, __bf16* __restrict__ Yb) {
    __shared__ float rs_[256], rss_[256];
    const int row = blockIdx.x;
    const int tid = threadIdx.x;
    const float* xr = X + (size_t)row * CE;
    const float* tr = T + (size_t)row * CE;

    float u[4], s = 0.0f, ss = 0.0f;
#pragma unroll
    for (int i = 0; i < 4; ++i) {
        int idx = tid + i * 256;
        u[i] = xr[idx] + tr[idx];
        s  += u[i];
        ss += u[i] * u[i];
    }
    rs_[tid] = s;
    rss_[tid] = ss;
    __syncthreads();
    for (int st = 128; st > 0; st >>= 1) {
        if (tid < st) {
            rs_[tid] += rs_[tid + st];
            rss_[tid] += rss_[tid + st];
        }
        __syncthreads();
    }
    float mean = rs_[0] * (1.0f / CE);
    float var  = rss_[0] * (1.0f / CE) - mean * mean;
    float inv  = rsqrtf(var + 1e-5f);
#pragma unroll
    for (int i = 0; i < 4; ++i) {
        int   idx = tid + i * 256;
        float y   = (u[i] - mean) * inv * g[idx] + be[idx];
        Yf[(size_t)row * CE + idx] = y;
        if constexpr (WRITE_BF16) Yb[(size_t)row * CE + idx] = (__bf16)y;
    }
}

// ---------------------------------------------------------------- launch ---
extern "C" void kernel_launch(void* const* d_in, const int* in_sizes, int n_in,
                              void* d_out, int out_size, void* d_ws,
                              size_t ws_size, hipStream_t stream) {
    const float* x       = (const float*)d_in[0];
    const float* Wq      = (const float*)d_in[1];
    const float* bq      = (const float*)d_in[2];
    const float* Wk      = (const float*)d_in[3];
    const float* bk      = (const float*)d_in[4];
    const float* Wv      = (const float*)d_in[5];
    const float* bv      = (const float*)d_in[6];
    const float* kernelW = (const float*)d_in[7];
    const float* Wo      = (const float*)d_in[8];
    const float* bo      = (const float*)d_in[9];
    const float* W1      = (const float*)d_in[10];
    const float* b1      = (const float*)d_in[11];
    const float* W2      = (const float*)d_in[12];
    const float* b2      = (const float*)d_in[13];
    const float* g1      = (const float*)d_in[14];
    const float* beta1   = (const float*)d_in[15];
    const float* g2      = (const float*)d_in[16];
    const float* beta2   = (const float*)d_in[17];

    char*  ws  = (char*)d_ws;
    size_t off = 0;
    auto alloc = [&](size_t bytes) -> void* {
        void* p = ws + off;
        off += (bytes + 255) & ~(size_t)255;
        return p;
    };

    __bf16* xb  = (__bf16*)alloc((size_t)CM * CE * 2);
    __bf16* Wqb = (__bf16*)alloc((size_t)CE * CE * 2);
    __bf16* Wkb = (__bf16*)alloc((size_t)CE * CE * 2);
    __bf16* Wvb = (__bf16*)alloc((size_t)CE * CE * 2);
    __bf16* Wob = (__bf16*)alloc((size_t)CE * CE * 2);
    __bf16* W1b = (__bf16*)alloc((size_t)CE * CFFN * 2);
    __bf16* W2b = (__bf16*)alloc((size_t)CFFN * CE * 2);
    __bf16* Qb  = (__bf16*)alloc((size_t)CM * CE * 2);
    __bf16* Kbb = (__bf16*)alloc((size_t)CM * CE * 2);
    __bf16* Vtb = (__bf16*)alloc((size_t)CM * CE * 2);
    __bf16* Ob  = (__bf16*)alloc((size_t)CM * CE * 2);
    float*  tmp = (float*)alloc((size_t)CM * CE * 4);
    float*  x1f = (float*)alloc((size_t)CM * CE * 4);
    __bf16* x1b = (__bf16*)alloc((size_t)CM * CE * 2);
    __bf16* hb  = (__bf16*)alloc((size_t)CM * CFFN * 2);

    auto cvt = [&](const float* src, __bf16* dst, int n) {
        cvt_f32_bf16_kernel<<<(n / 4 + 255) / 256, 256, 0, stream>>>(src, dst, n);
    };
    cvt(x, xb, CM * CE);
    cvt(Wq, Wqb, CE * CE);
    cvt(Wk, Wkb, CE * CE);
    cvt(Wv, Wvb, CE * CE);
    cvt(Wo, Wob, CE * CE);
    cvt(W1, W1b, CE * CFFN);
    cvt(W2, W2b, CFFN * CE);

    dim3 blk(256);
    dim3 gP(CE / 128, CM / 128);     // 8 x 32
    dim3 gF1(CFFN / 128, CM / 128);  // 32 x 32

    // Q/K/V projections (Q folds kernelW scale; V written transposed per head)
    gemm_bf16_kernel<EPI_QKV_SCALED><<<gP, blk, 0, stream>>>(
        xb, Wqb, bq, kernelW, Qb, CM, CE, CE, CS);
    gemm_bf16_kernel<EPI_QKV><<<gP, blk, 0, stream>>>(
        xb, Wkb, bk, nullptr, Kbb, CM, CE, CE, CS);
    gemm_bf16_kernel<EPI_VT><<<gP, blk, 0, stream>>>(
        xb, Wvb, bv, nullptr, Vtb, CM, CE, CE, CS);

    // flash attention
    attn_kernel<<<CB * CH * (CS / 128), blk, 0, stream>>>(Qb, Kbb, Vtb, Ob, CS);

    // output projection, residual + LN1
    gemm_bf16_kernel<EPI_F32><<<gP, blk, 0, stream>>>(
        Ob, Wob, bo, nullptr, tmp, CM, CE, CE, CS);
    resid_ln_kernel<true><<<CM, blk, 0, stream>>>(x, tmp, g1, beta1, x1f, x1b);

    // FFN
    gemm_bf16_kernel<EPI_BF16_RELU><<<gF1, blk, 0, stream>>>(
        x1b, W1b, b1, nullptr, hb, CM, CFFN, CE, CS);
    gemm_bf16_kernel<EPI_F32><<<gP, blk, 0, stream>>>(
        hb, W2b, b2, nullptr, tmp, CM, CE, CFFN, CS);

    // residual + LN2 -> output
    resid_ln_kernel<false><<<CM, blk, 0, stream>>>(x1f, tmp, g2, beta2,
                                                   (float*)d_out, nullptr);
}
